// ConditionGateAttention_85641647882454
// MI455X (gfx1250) — matrix operand order repaired and checked by reading, not compiled, not run on hardware
//
#include <hip/hip_runtime.h>
#include <math.h>
#include <stdint.h>

// ---------------------------------------------------------------------------
// ConditionGateAttention for MI455X (gfx1250, wave32, WMMA 16x16x32 f16,
// async global->LDS staging). B=2, T=2048, M=77, C=512, H=8, D=64.
// ---------------------------------------------------------------------------

typedef __attribute__((ext_vector_type(16))) _Float16 v16h;
typedef __attribute__((ext_vector_type(8)))  _Float16 h8;
typedef __attribute__((ext_vector_type(4)))  _Float16 h4;
typedef __attribute__((ext_vector_type(8)))  float    v8f;

#define WMMA16(a, b, c) \
  __builtin_amdgcn_wmma_f32_16x16x32_f16(false, (a), false, (b), (short)0, (c), false, false)

// Async copy of 16 bytes global -> LDS (per lane), tracked by ASYNCcnt.
__device__ __forceinline__ void async_b128(unsigned lds_off, const _Float16* g) {
  unsigned long long ga = (unsigned long long)(uintptr_t)g;
  asm volatile("global_load_async_to_lds_b128 %0, %1, off"
               :: "v"(lds_off), "v"(ga) : "memory");
}
#define LDSOFF(p) ((unsigned)(uintptr_t)(p))
#define WAIT_ASYNC_0() asm volatile("s_wait_asynccnt 0x0" ::: "memory")
#define WAIT_ASYNC_4() asm volatile("s_wait_asynccnt 0x4" ::: "memory")

// A-fragment (16x32, MxK): lane l holds row m=l&15; g=l>>4:
//   elems 0..7 -> K=g*8+0..7 ; elems 8..15 -> K=16+g*8+0..7
__device__ __forceinline__ v16h load_afrag(const _Float16* row, int g) {
  h8 lo = *(const h8*)(row + g * 8);
  h8 hi = *(const h8*)(row + 16 + g * 8);
  v16h r;
#pragma unroll
  for (int i = 0; i < 8; i++) { r[i] = lo[i]; r[i + 8] = hi[i]; }
  return r;
}
// B-fragment (32x16, KxN): lane l holds col n=l&15; g=l>>4:
//   elems 0..7 -> K=g*8+0..7 ; elems 8..15 -> K=16+g*8+0..7
// `row` points at a K-contiguous column (tile stored transposed in LDS).
__device__ __forceinline__ v16h load_bfrag(const _Float16* row, int g) {
  h8 lo = *(const h8*)(row + g * 8);
  h8 hi = *(const h8*)(row + 16 + g * 8);
  v16h r;
#pragma unroll
  for (int i = 0; i < 8; i++) { r[i] = lo[i]; r[i + 8] = hi[i]; }
  return r;
}

// ---------------------------------------------------------------------------
// Pre-passes: f32->f16 convert, and 512x512 transpose+convert for weights.
// ---------------------------------------------------------------------------
__global__ __launch_bounds__(256) void cvt_h(const float* __restrict__ in,
                                             _Float16* __restrict__ out, int n4) {
  int i = blockIdx.x * 256 + threadIdx.x;
  if (i < n4) {
    float4 f = ((const float4*)in)[i];
    h4 o = {(_Float16)f.x, (_Float16)f.y, (_Float16)f.z, (_Float16)f.w};
    ((h4*)out)[i] = o;
  }
}

__global__ __launch_bounds__(256) void transpose_cvt(const float* __restrict__ in,
                                                     _Float16* __restrict__ out) {
  __shared__ float t[32][33];
  int k0 = blockIdx.x * 32, n0 = blockIdx.y * 32;
  int tx = threadIdx.x & 31, ty = threadIdx.x >> 5;  // 32 x 8
#pragma unroll
  for (int i = 0; i < 32; i += 8)
    t[ty + i][tx] = in[(size_t)(k0 + ty + i) * 512 + n0 + tx];
  __syncthreads();
#pragma unroll
  for (int i = 0; i < 32; i += 8)
    out[(size_t)(n0 + ty + i) * 512 + k0 + tx] = (_Float16)t[tx][ty + i];
}

// ---------------------------------------------------------------------------
// GEMM: out[N,512] = A[N,512](f16) @ W(pre-transposed Wt[n][k], f16) + bias.
// 128x128 tile / 256 threads; double-buffered async global->LDS staging.
// Out-of-range rows are clamped (their outputs are never stored).
// ---------------------------------------------------------------------------
template <bool OUTF16>
__global__ __launch_bounds__(256) void gemm512h(const _Float16* __restrict__ A,
                                                const _Float16* __restrict__ Wt,
                                                const float* __restrict__ bias,
                                                void* __restrict__ outp, int Nrows) {
  __shared__ __attribute__((aligned(16))) _Float16 As[2][128][40];
  __shared__ __attribute__((aligned(16))) _Float16 Ws[2][128][40];
  const int tid = threadIdx.x;
  const int wave = tid >> 5, lane = tid & 31;
  const int ln = lane & 15, g = lane >> 4;
  const int row0 = blockIdx.x * 128;
  const int col0 = blockIdx.y * 128;
  const int wm = (wave >> 1) * 32;
  const int wn = (wave & 1) * 64;

  auto issue = [&](int ks, int buf) {
#pragma unroll
    for (int i = 0; i < 2; i++) {
      int idx = tid + i * 256;        // 0..511
      int r = idx >> 2, c = idx & 3;  // row, 16B chunk (4 per 64B row)
      int ar = row0 + r; if (ar > Nrows - 1) ar = Nrows - 1;
      async_b128(LDSOFF(&As[buf][r][c * 8]), A  + (size_t)ar * 512 + ks * 32 + c * 8);
      async_b128(LDSOFF(&Ws[buf][r][c * 8]), Wt + (size_t)(col0 + r) * 512 + ks * 32 + c * 8);
    }
  };

  const v8f zf = {0.f, 0.f, 0.f, 0.f, 0.f, 0.f, 0.f, 0.f};
  v8f acc[2][4];
#pragma unroll
  for (int i = 0; i < 2; i++)
#pragma unroll
    for (int j = 0; j < 4; j++) acc[i][j] = zf;

  issue(0, 0);
  for (int ks = 0; ks < 16; ks++) {
    int cur = ks & 1;
    if (ks < 15) { issue(ks + 1, cur ^ 1); WAIT_ASYNC_4(); }
    else         { WAIT_ASYNC_0(); }
    __syncthreads();

    v16h af[2], bf[4];
#pragma unroll
    for (int i = 0; i < 2; i++) af[i] = load_afrag(&As[cur][wm + i * 16 + ln][0], g);
#pragma unroll
    for (int j = 0; j < 4; j++) bf[j] = load_bfrag(&Ws[cur][wn + j * 16 + ln][0], g);
#pragma unroll
    for (int i = 0; i < 2; i++)
#pragma unroll
      for (int j = 0; j < 4; j++) acc[i][j] = WMMA16(af[i], bf[j], acc[i][j]);
    __syncthreads();
  }

#pragma unroll
  for (int i = 0; i < 2; i++)
#pragma unroll
    for (int j = 0; j < 4; j++) {
      int col = col0 + wn + j * 16 + ln;
      float bv = bias[col];
#pragma unroll
      for (int r = 0; r < 8; r++) {
        int rowg = row0 + wm + i * 16 + g * 8 + r;
        if (rowg < Nrows) {
          float val = acc[i][j][r] + bv;
          if (OUTF16) ((_Float16*)outp)[(size_t)rowg * 512 + col] = (_Float16)val;
          else        ((float*)outp)[(size_t)rowg * 512 + col] = val;
        }
      }
    }
}

// ---------------------------------------------------------------------------
// Causal flash self-attention, f16 q/k/v. Grid (B*H, T/128), block 256.
// Writes y (fp32, for gating math) and yh (f16, A-operand of gate GEMM).
// ---------------------------------------------------------------------------
__global__ __launch_bounds__(256) void self_attn_h(const _Float16* __restrict__ q,
                                                   const _Float16* __restrict__ k,
                                                   const _Float16* __restrict__ v,
                                                   float* __restrict__ y,
                                                   _Float16* __restrict__ yh) {
  __shared__ __attribute__((aligned(16))) _Float16 Qs[128][72];
  __shared__ __attribute__((aligned(16))) _Float16 Ks[64][72];
  __shared__ __attribute__((aligned(16))) _Float16 Vt[64][72];     // [d][key]
  __shared__ __attribute__((aligned(16))) _Float16 Ps[8][16][72];  // per-wave P
  const int tid = threadIdx.x, wave = tid >> 5, lane = tid & 31;
  const int ln = lane & 15, g = lane >> 4;
  const int bh = blockIdx.x, b = bh >> 3, h = bh & 7;
  const int qrow0 = blockIdx.y * 128;
  const size_t base = (size_t)b * 2048 * 512 + h * 64;

  // Q tile async: 128 rows x 8 chunks of 16B
#pragma unroll
  for (int i = 0; i < 4; i++) {
    int idx = tid + i * 256;
    int r = idx >> 3, c = idx & 7;
    async_b128(LDSOFF(&Qs[r][c * 8]), q + base + (size_t)(qrow0 + r) * 512 + c * 8);
  }
  WAIT_ASYNC_0();
  __syncthreads();

  v16h qf[2];
  qf[0] = load_afrag(&Qs[wave * 16 + ln][0], g);
  qf[1] = load_afrag(&Qs[wave * 16 + ln][32], g);

  const v8f zf = {0.f, 0.f, 0.f, 0.f, 0.f, 0.f, 0.f, 0.f};
  v8f o[4];
#pragma unroll
  for (int j = 0; j < 4; j++) o[j] = zf;
  float mrow[8], lrow[8];
#pragma unroll
  for (int r = 0; r < 8; r++) { mrow[r] = -1e30f; lrow[r] = 0.f; }

  const int vd = tid & 63, vsg = tid >> 6;  // V-transpose staging role
  const int nkt = (qrow0 >> 6) + 2;
  for (int kt = 0; kt < nkt; kt++) {
    // K tile async (row-major)
#pragma unroll
    for (int i = 0; i < 2; i++) {
      int idx = tid + i * 256;
      int s = idx >> 3, c = idx & 7;
      async_b128(LDSOFF(&Ks[s][c * 8]), k + base + (size_t)(kt * 64 + s) * 512 + c * 8);
    }
    // V tile transposed: Vt[d][s]; coalesced u16 gathers, b128 LDS stores
#pragma unroll
    for (int j = 0; j < 2; j++) {
      int s0 = vsg * 16 + j * 8;
      h8 tmp;
#pragma unroll
      for (int u = 0; u < 8; u++) tmp[u] = v[base + (size_t)(kt * 64 + s0 + u) * 512 + vd];
      *(h8*)&Vt[vd][s0] = tmp;
    }
    WAIT_ASYNC_0();
    __syncthreads();

    // S = Q K^T (16 rows x 64 keys per wave), scale + causal mask
    float sv[4][8];
#pragma unroll
    for (int ct = 0; ct < 4; ct++) {
      v8f s = zf;
      s = WMMA16(qf[0], load_bfrag(&Ks[ct * 16 + ln][0], g), s);
      s = WMMA16(qf[1], load_bfrag(&Ks[ct * 16 + ln][32], g), s);
      int sg = kt * 64 + ct * 16 + ln;
#pragma unroll
      for (int r = 0; r < 8; r++) {
        int tg = qrow0 + wave * 16 + g * 8 + r;
        sv[ct][r] = (sg > tg) ? -1e30f : s[r] * 0.125f;
      }
    }

    // Online softmax (row reductions across 16-lane half-groups)
#pragma unroll
    for (int r = 0; r < 8; r++) {
      float mloc = sv[0][r];
#pragma unroll
      for (int ct = 1; ct < 4; ct++) mloc = fmaxf(mloc, sv[ct][r]);
#pragma unroll
      for (int off = 1; off < 16; off <<= 1) mloc = fmaxf(mloc, __shfl_xor(mloc, off, 32));
      float mnew = fmaxf(mrow[r], mloc);
      float alpha = __expf(mrow[r] - mnew);
      float lsum = 0.f;
#pragma unroll
      for (int ct = 0; ct < 4; ct++) {
        float p = __expf(sv[ct][r] - mnew);
        sv[ct][r] = p;
        lsum += p;
      }
#pragma unroll
      for (int off = 1; off < 16; off <<= 1) lsum += __shfl_xor(lsum, off, 32);
      lrow[r] = lrow[r] * alpha + lsum;
      mrow[r] = mnew;
#pragma unroll
      for (int j = 0; j < 4; j++) o[j][r] *= alpha;
    }

    // Stage P then O += P @ V
#pragma unroll
    for (int ct = 0; ct < 4; ct++)
#pragma unroll
      for (int r = 0; r < 8; r++) Ps[wave][g * 8 + r][ct * 16 + ln] = (_Float16)sv[ct][r];
#pragma unroll
    for (int kk = 0; kk < 2; kk++) {
      v16h pf = load_afrag(&Ps[wave][ln][kk * 32], g);
#pragma unroll
      for (int j = 0; j < 4; j++)
        o[j] = WMMA16(pf, load_bfrag(&Vt[j * 16 + ln][kk * 32], g), o[j]);
    }
    __syncthreads();
  }

#pragma unroll
  for (int j = 0; j < 4; j++)
#pragma unroll
    for (int r = 0; r < 8; r++) {
      int tg = qrow0 + wave * 16 + g * 8 + r;
      float val = o[j][r] / lrow[r];
      size_t idx = base + (size_t)tg * 512 + j * 16 + ln;
      y[idx] = val;
      yh[idx] = (_Float16)val;
    }
}

// ---------------------------------------------------------------------------
// Cross-attention (M=77 keys padded to 96; clamped rows are masked to P=0).
// Grid (B*H, T/64), block 128. Writes yc (fp32) and ych (f16).
// ---------------------------------------------------------------------------
__global__ __launch_bounds__(128) void cross_attn_h(const _Float16* __restrict__ q,
                                                    const _Float16* __restrict__ kc,
                                                    const _Float16* __restrict__ vc,
                                                    const int* __restrict__ pmask,
                                                    float* __restrict__ yc,
                                                    _Float16* __restrict__ ych) {
  __shared__ __attribute__((aligned(16))) _Float16 Qs[64][72];
  __shared__ __attribute__((aligned(16))) _Float16 Ks[96][72];
  __shared__ __attribute__((aligned(16))) _Float16 Vt[64][104];
  __shared__ __attribute__((aligned(16))) _Float16 Ps[4][16][104];
  const int tid = threadIdx.x, wave = tid >> 5, lane = tid & 31;
  const int ln = lane & 15, g = lane >> 4;
  const int bh = blockIdx.x, b = bh >> 3, h = bh & 7;
  const int qrow0 = blockIdx.y * 64;
  const size_t qbase = (size_t)b * 2048 * 512 + h * 64;
  const size_t cbase = (size_t)b * 77 * 512 + h * 64;

#pragma unroll
  for (int i = 0; i < 4; i++) {
    int idx = tid + i * 128;
    int r = idx >> 3, c = idx & 7;
    async_b128(LDSOFF(&Qs[r][c * 8]), q + qbase + (size_t)(qrow0 + r) * 512 + c * 8);
  }
#pragma unroll
  for (int i = 0; i < 6; i++) {
    int idx = tid + i * 128;
    int s = idx >> 3, c = idx & 7;
    int sc = (s > 76) ? 76 : s;  // clamped rows masked later
    async_b128(LDSOFF(&Ks[s][c * 8]), kc + cbase + (size_t)sc * 512 + c * 8);
  }
  {
    const int vd = tid & 63, vsg = tid >> 6;  // 2 s-groups of 48
#pragma unroll
    for (int j = 0; j < 6; j++) {
      int s0 = vsg * 48 + j * 8;
      h8 tmp;
#pragma unroll
      for (int u = 0; u < 8; u++) {
        int sc = (s0 + u > 76) ? 76 : (s0 + u);
        tmp[u] = vc[cbase + (size_t)sc * 512 + vd];
      }
      *(h8*)&Vt[vd][s0] = tmp;
    }
  }
  WAIT_ASYNC_0();
  __syncthreads();

  v16h qf0 = load_afrag(&Qs[wave * 16 + ln][0], g);
  v16h qf1 = load_afrag(&Qs[wave * 16 + ln][32], g);

  const v8f zf = {0.f, 0.f, 0.f, 0.f, 0.f, 0.f, 0.f, 0.f};
  float sv[6][8];
#pragma unroll
  for (int ct = 0; ct < 6; ct++) {
    v8f s = zf;
    s = WMMA16(qf0, load_bfrag(&Ks[ct * 16 + ln][0], g), s);
    s = WMMA16(qf1, load_bfrag(&Ks[ct * 16 + ln][32], g), s);
    int col = ct * 16 + ln;
    bool valid = (col < 77) && (pmask[b * 77 + ((col < 77) ? col : 0)] != 0);
#pragma unroll
    for (int r = 0; r < 8; r++) sv[ct][r] = valid ? s[r] * 0.125f : -1e30f;
  }

  float lrow[8];
#pragma unroll
  for (int r = 0; r < 8; r++) {
    float m = sv[0][r];
#pragma unroll
    for (int ct = 1; ct < 6; ct++) m = fmaxf(m, sv[ct][r]);
#pragma unroll
    for (int off = 1; off < 16; off <<= 1) m = fmaxf(m, __shfl_xor(m, off, 32));
    float lsum = 0.f;
#pragma unroll
    for (int ct = 0; ct < 6; ct++) {
      float p = __expf(sv[ct][r] - m);
      sv[ct][r] = p;
      lsum += p;
    }
#pragma unroll
    for (int off = 1; off < 16; off <<= 1) lsum += __shfl_xor(lsum, off, 32);
    lrow[r] = lsum;
  }

#pragma unroll
  for (int ct = 0; ct < 6; ct++)
#pragma unroll
    for (int r = 0; r < 8; r++) Ps[wave][g * 8 + r][ct * 16 + ln] = (_Float16)sv[ct][r];

  v8f o[4];
#pragma unroll
  for (int j = 0; j < 4; j++) o[j] = zf;
#pragma unroll
  for (int kk = 0; kk < 3; kk++) {
    v16h pf = load_afrag(&Ps[wave][ln][kk * 32], g);
#pragma unroll
    for (int j = 0; j < 4; j++)
      o[j] = WMMA16(pf, load_bfrag(&Vt[j * 16 + ln][kk * 32], g), o[j]);
  }

#pragma unroll
  for (int j = 0; j < 4; j++)
#pragma unroll
    for (int r = 0; r < 8; r++) {
      int tg = qrow0 + wave * 16 + g * 8 + r;
      float val = o[j][r] / lrow[r];
      size_t idx = qbase + (size_t)tg * 512 + j * 16 + ln;
      yc[idx] = val;
      ych[idx] = (_Float16)val;
    }
}

// ---------------------------------------------------------------------------
// fused_h = f16( sigmoid(g1)*yc + sigmoid(g2)*y )
// ---------------------------------------------------------------------------
__global__ __launch_bounds__(256) void gate_combine(const float* __restrict__ g1,
                                                    const float* __restrict__ g2,
                                                    const float* __restrict__ y,
                                                    const float* __restrict__ yc,
                                                    _Float16* __restrict__ fused, int n) {
  int i = blockIdx.x * 256 + threadIdx.x;
  if (i < n) {
    float gs = 1.f / (1.f + __expf(-g1[i]));
    float gc = 1.f / (1.f + __expf(-g2[i]));
    fused[i] = (_Float16)(gs * yc[i] + gc * y[i]);
  }
}

// ---------------------------------------------------------------------------
extern "C" void kernel_launch(void* const* d_in, const int* in_sizes, int n_in,
                              void* d_out, int out_size, void* d_ws, size_t ws_size,
                              hipStream_t stream) {
  (void)in_sizes; (void)n_in; (void)out_size; (void)ws_size;
  const float* x = (const float*)d_in[0];
  const float* c = (const float*)d_in[1];
  // d_in[2]: attn_mask (causal tril) -- implemented analytically
  const int* pmask = (const int*)d_in[3];
  const float* Wf[8] = {(const float*)d_in[4],  (const float*)d_in[6],
                        (const float*)d_in[8],  (const float*)d_in[10],
                        (const float*)d_in[12], (const float*)d_in[14],
                        (const float*)d_in[16], (const float*)d_in[18]};
  const float* bq  = (const float*)d_in[5];
  const float* bk  = (const float*)d_in[7];
  const float* bv  = (const float*)d_in[9];
  const float* bkc = (const float*)d_in[11];
  const float* bvc = (const float*)d_in[13];
  const float* bg1 = (const float*)d_in[15];
  const float* bg2 = (const float*)d_in[17];
  const float* bp  = (const float*)d_in[19];
  float* out = (float*)d_out;

  // ---- workspace carve-up (bytes, 256-aligned) -------------------------
  size_t o = 0;
  char* wsb = (char*)d_ws;
  auto alloc = [&](size_t bytes) -> void* {
    void* p = wsb + o;
    o += (bytes + 255) & ~(size_t)255;
    return p;
  };
  const size_t NT = (size_t)4096 * 512;   // B*T*C
  const size_t NM = (size_t)154 * 512;    // B*M*C
  _Float16* x_h = (_Float16*)alloc(NT * 2);
  _Float16* c_h = (_Float16*)alloc(NM * 2);
  _Float16* Wt[8];
  for (int i = 0; i < 8; i++) Wt[i] = (_Float16*)alloc((size_t)512 * 512 * 2);
  _Float16* q_h  = (_Float16*)alloc(NT * 2);
  _Float16* k_h  = (_Float16*)alloc(NT * 2);
  _Float16* v_h  = (_Float16*)alloc(NT * 2);
  _Float16* kc_h = (_Float16*)alloc(NM * 2);
  _Float16* vc_h = (_Float16*)alloc(NM * 2);
  float*    y    = (float*)alloc(NT * 4);
  float*    yc   = (float*)alloc(NT * 4);
  _Float16* y_h  = (_Float16*)alloc(NT * 2);
  _Float16* yc_h = (_Float16*)alloc(NT * 2);
  float*    g1   = (float*)alloc(NT * 4);
  float*    g2   = (float*)alloc(NT * 4);
  _Float16* fu_h = (_Float16*)alloc(NT * 2);

  const dim3 blk(256);
  // ---- pre-convert inputs & weights ------------------------------------
  for (int i = 0; i < 8; i++)
    transpose_cvt<<<dim3(16, 16), blk, 0, stream>>>(Wf[i], Wt[i]);
  cvt_h<<<dim3((int)(NT / 4 / 256)), blk, 0, stream>>>(x, x_h, (int)(NT / 4));
  cvt_h<<<dim3((int)((NM / 4 + 255) / 256)), blk, 0, stream>>>(c, c_h, (int)(NM / 4));

  // ---- projections (f16 out) -------------------------------------------
  gemm512h<true><<<dim3(32, 4), blk, 0, stream>>>(x_h, Wt[0], bq,  q_h,  4096);
  gemm512h<true><<<dim3(32, 4), blk, 0, stream>>>(x_h, Wt[1], bk,  k_h,  4096);
  gemm512h<true><<<dim3(32, 4), blk, 0, stream>>>(x_h, Wt[2], bv,  v_h,  4096);
  gemm512h<true><<<dim3(2, 4),  blk, 0, stream>>>(c_h, Wt[3], bkc, kc_h, 154);
  gemm512h<true><<<dim3(2, 4),  blk, 0, stream>>>(c_h, Wt[4], bvc, vc_h, 154);

  // ---- attention ---------------------------------------------------------
  self_attn_h<<<dim3(16, 16), blk, 0, stream>>>(q_h, k_h, v_h, y, y_h);
  cross_attn_h<<<dim3(16, 32), dim3(128), 0, stream>>>(q_h, kc_h, vc_h, pmask, yc, yc_h);

  // ---- gating + output projection ---------------------------------------
  gemm512h<false><<<dim3(32, 4), blk, 0, stream>>>(y_h,  Wt[5], bg1, g1, 4096);
  gemm512h<false><<<dim3(32, 4), blk, 0, stream>>>(yc_h, Wt[6], bg2, g2, 4096);
  gate_combine<<<dim3((int)(NT / 256)), blk, 0, stream>>>(g1, g2, y, yc, fu_h, (int)NT);
  gemm512h<false><<<dim3(32, 4), blk, 0, stream>>>(fu_h, Wt[7], bp, out, 4096);
}
